// KNNPathlineTransformerLayer_59708635349033
// MI455X (gfx1250) — compile-verified, matrix operand order checked by reading, not run on hardware
//
#include <hip/hip_runtime.h>
#include <hip/hip_bf16.h>

typedef __attribute__((ext_vector_type(16))) _Float16 v16h;
typedef __attribute__((ext_vector_type(8)))  _Float16 v8h;
typedef __attribute__((ext_vector_type(8)))  float    v8f;

// __builtin_amdgcn_cvt_pkrtz returns __fp16-element vectors; keep that type for packing
typedef __attribute__((ext_vector_type(2))) __fp16 v2hf;
typedef __attribute__((ext_vector_type(4))) __fp16 v4hf;
typedef __attribute__((ext_vector_type(8))) __fp16 v8hf;

#define BN    65536   // B*N
#define NPTS  16384   // N
#define CH    64
#define KNN   16
#define WAVES 8

// d_ws layout (halves): 8 matrices, each stored row-major [64 x 64]
#define WQ_OFF   0
#define WK_OFF   4096
#define WV_OFF   8192
#define PW2_OFF  12288
#define AW1_OFF  16384
#define AW2_OFF  20480
#define OW_OFF   24576
#define WSPL_OFF 28672    // combined KAN weight, K padded 27->64 (only cols 0..31 ever read)
#define WS_HALVES 32768

__device__ __forceinline__ v8f wmma_f16(v16h a, v16h b, v8f c) {
  // D = A(16x32 f16) * B(32x16 f16) + C(16x16 f32)
  return __builtin_amdgcn_wmma_f32_16x16x32_f16(false, a, false, b, (short)0, c, false, false);
}

__device__ __forceinline__ v16h cat8(v8h lo, v8h hi) {
  return __builtin_shufflevector(lo, hi, 0, 1, 2, 3, 4, 5, 6, 7, 8, 9, 10, 11, 12, 13, 14, 15);
}

// pack 8 floats -> v8h with v_cvt_pk_rtz_f16_f32
__device__ __forceinline__ v8h pk8(float4 f0, float4 f1) {
  v2hf p0 = __builtin_amdgcn_cvt_pkrtz(f0.x, f0.y);
  v2hf p1 = __builtin_amdgcn_cvt_pkrtz(f0.z, f0.w);
  v2hf p2 = __builtin_amdgcn_cvt_pkrtz(f1.x, f1.y);
  v2hf p3 = __builtin_amdgcn_cvt_pkrtz(f1.z, f1.w);
  v4hf q0 = __builtin_shufflevector(p0, p1, 0, 1, 2, 3);
  v4hf q1 = __builtin_shufflevector(p2, p3, 0, 1, 2, 3);
  v8hf r  = __builtin_shufflevector(q0, q1, 0, 1, 2, 3, 4, 5, 6, 7);
  return __builtin_bit_cast(v8h, r);
}

// A-fragment (16x32 f16) from LDS row-major [16 x stride] halves (stride 0 => broadcast row).
// lane<16: M=lane, K = 32c + {0..7,16..23}; lane>=16: M=lane-16, K = 32c + {8..15,24..31}
__device__ __forceinline__ v16h load_a_lds(const _Float16* base, int stride, int chunk, int lane) {
  const int m  = lane & 15;
  const int kb = chunk * 32 + ((lane >> 4) << 3);
  const _Float16* p = base + m * stride + kb;
  return cat8(*(const v8h*)p, *(const v8h*)(p + 16));
}

// B-fragment (32x16 f16) for D = A * W^T, W pre-converted f16 row-major [64 x 64] in global.
// B[k][n] = W[tile*16+n][k] -> lane n reads two contiguous 16B groups of one weight row.
__device__ __forceinline__ v16h load_b_gh(const _Float16* __restrict__ Wh, int tile, int chunk, int lane) {
  const int col = (lane & 15) + tile * 16;
  const int kb  = chunk * 32 + ((lane >> 4) << 3);
  const _Float16* p = Wh + col * CH + kb;
  return cat8(*(const v8h*)p, *(const v8h*)(p + 16));
}

__device__ __forceinline__ float silu_f(float u) { return u / (1.0f + __expf(-u)); }

// Uniform cubic B-spline (GRID_SIZE=5, SPLINE_ORDER=3), knots t_j=(j-3)*0.4-1, j=0..11.
// For u in interval i (t_i <= u < t_{i+1}) only bases i-3..i are nonzero, with the standard
// uniform-knot closed form. rel in (-1,1) => i in [3,7], so all 4 live bases land in o[0..7].
__device__ __forceinline__ void spline_bases(float u, float* o) {
  const float s  = (u + 2.2f) * 2.5f;   // knot-space coordinate
  const float fi = floorf(s);
  const int   i  = (int)fi;
  const float f  = s - fi;
  const float omf = 1.0f - f;
  const float f2 = f * f, f3 = f2 * f;
  const float c6 = 1.0f / 6.0f;
  const float B0 = omf * omf * omf * c6;
  const float B1 = (3.0f * f3 - 6.0f * f2 + 4.0f) * c6;
  const float B2 = (-3.0f * f3 + 3.0f * f2 + 3.0f * f + 1.0f) * c6;
  const float B3 = f3 * c6;
#pragma unroll
  for (int j = 0; j < 8; ++j) {
    float v = 0.0f;
    v = (j == i - 3) ? B0 : v;
    v = (j == i - 2) ? B1 : v;
    v = (j == i - 1) ? B2 : v;
    v = (j == i)     ? B3 : v;
    o[j] = v;
  }
}

// ---------------- prep kernel: convert weights to f16 in d_ws (once per launch) -------------
__global__ __launch_bounds__(256)
void prep_weights(const float* __restrict__ wq, const float* __restrict__ wk,
                  const float* __restrict__ wv, const float* __restrict__ pos_w2,
                  const float* __restrict__ aw1, const float* __restrict__ aw2,
                  const float* __restrict__ out_w,
                  const float* __restrict__ base_w, const float* __restrict__ spline_w,
                  const float* __restrict__ scaler,
                  _Float16* __restrict__ ws) {
  const int blk = blockIdx.x, t = threadIdx.x;
  if (blk < 7) {
    const float* srcs[7] = {wq, wk, wv, pos_w2, aw1, aw2, out_w};
    const float* src = srcs[blk];
    _Float16* dst = ws + blk * 4096;
#pragma unroll 4
    for (int i = t; i < 4096; i += 256) dst[i] = (_Float16)src[i];
  } else {
    // combined KAN weight, padded [64 x 64]: row c = [base_w(3) | spline_w*scaler(24) | zeros]
    _Float16* dst = ws + WSPL_OFF;
#pragma unroll 4
    for (int i = t; i < 4096; i += 256) {
      const int c = i >> 6, j = i & 63;
      float v = 0.0f;
      if (j < 3) {
        v = base_w[c * 3 + j];
      } else if (j < 27) {
        const int q = j - 3, d = q >> 3, g = q & 7;
        v = spline_w[(c * 3 + d) * 8 + g] * scaler[c * 3 + d];
      }
      dst[i] = (_Float16)v;
    }
  }
}

// ---------------- main kernel: one wave per query point, 8 waves/block -------------
__global__ __launch_bounds__(256)
void kan_knn_transformer(const float* __restrict__ x,
                         const float* __restrict__ pos,
                         const int*   __restrict__ knn_idx,
                         const float* __restrict__ pos_b2,
                         const float* __restrict__ attn_b1,
                         const float* __restrict__ attn_b2,
                         const float* __restrict__ out_b,
                         const _Float16* __restrict__ ws,
                         float* __restrict__ out)
{
  __shared__ _Float16 sFeat[WAVES][KNN * CH];  // gathered neighbor features (f16)
  __shared__ _Float16 sB[WAVES][KNN * CH];     // staging: spl(16x32, aliased) -> h -> energy -> t1
  __shared__ _Float16 sXh[WAVES][CH];          // x[p] as halves (broadcast A for q)
  __shared__ _Float16 sOh[WAVES][CH];          // attention output as halves (broadcast A for y)

  const int tid  = threadIdx.x;
  const int w    = tid >> 5;
  const int lane = tid & 31;
  const int p    = blockIdx.x * WAVES + w;   // global point index (b*N + n)
  const int b    = p >> 14;                  // N = 16384
  const int lo   = lane & 15;
  const int hi   = lane >> 4;

  // warm the 64KB of converted weights into cache (CDNA5 prefetch path)
  if (tid < 32) __builtin_prefetch(ws + tid * 1024, 0, 3);

  _Float16* sBw = sB[w];

  // ---- stage x[p] as halves ----
  sXh[w][lane]      = (_Float16)x[(size_t)p * CH + lane];
  sXh[w][lane + 32] = (_Float16)x[(size_t)p * CH + lane + 32];

  // ---- gather knn features -> LDS f16 (lane: neighbor = lane>>1, channel half = (lane&1)*32) ----
  {
    const int nbr = knn_idx[(size_t)p * KNN + (lane >> 1)];
    const float* src = x + ((size_t)(b * NPTS + nbr)) * CH + (lane & 1) * 32;
    _Float16* dst = &sFeat[w][(lane >> 1) * CH + (lane & 1) * 32];
#pragma unroll
    for (int j = 0; j < 32; j += 8) {
      float4 f0 = *(const float4*)(src + j);
      float4 f1 = *(const float4*)(src + j + 4);
      *(v8h*)(dst + j) = pk8(f0, f1);
    }
  }

  // ---- spline features (lanes 0..15, one neighbor each) -> sB as [16 x 32] ----
  if (lane < KNN) {
    const int nbr = knn_idx[(size_t)p * KNN + lane];
    float rel[3];
#pragma unroll
    for (int d = 0; d < 3; ++d)
      rel[d] = pos[(size_t)(b * NPTS + nbr) * 3 + d] - pos[(size_t)p * 3 + d];
    float vals[32];
#pragma unroll
    for (int d = 0; d < 3; ++d) vals[d] = silu_f(rel[d]);
#pragma unroll
    for (int d = 0; d < 3; ++d) spline_bases(rel[d], &vals[3 + d * 8]);
#pragma unroll
    for (int j = 27; j < 32; ++j) vals[j] = 0.0f;
    _Float16* row = &sBw[lane * 32];
#pragma unroll
    for (int j = 0; j < 32; j += 8) {
      float4 f0 = {vals[j + 0], vals[j + 1], vals[j + 2], vals[j + 3]};
      float4 f1 = {vals[j + 4], vals[j + 5], vals[j + 6], vals[j + 7]};
      *(v8h*)(row + j) = pk8(f0, f1);
    }
  }
  __builtin_amdgcn_sched_barrier(0);

  // ---- q = x @ wq^T via broadcast-A WMMA (every D row == q) ----
  v8f Q[4];
  {
    const v16h ax0 = load_a_lds(sXh[w], 0, 0, lane);
    const v16h ax1 = load_a_lds(sXh[w], 0, 1, lane);
#pragma unroll
    for (int t = 0; t < 4; ++t) {
      v8f acc = {};
      acc = wmma_f16(ax0, load_b_gh(ws + WQ_OFF, t, 0, lane), acc);
      acc = wmma_f16(ax1, load_b_gh(ws + WQ_OFF, t, 1, lane), acc);
      Q[t] = acc;
    }
  }

  // ---- h = relu(spl(16x32) @ Wspl^T) -> sB [16 x 64] (overwrites spl region) ----
  {
    const v16h a = load_a_lds(sBw, 32, 0, lane);
    __builtin_amdgcn_sched_barrier(0);
#pragma unroll
    for (int t = 0; t < 4; ++t) {
      v8f acc = {};
      acc = wmma_f16(a, load_b_gh(ws + WSPL_OFF, t, 0, lane), acc);  // Wspl padded to 64-wide rows
#pragma unroll
      for (int r = 0; r < 8; ++r)
        sBw[(r + 8 * hi) * CH + t * 16 + lo] = (_Float16)fmaxf(acc[r], 0.0f);
    }
  }
  __builtin_amdgcn_sched_barrier(0);

  // ---- pos_enc = h @ pos_w2^T + pos_b2 (registers) ----
  v8f P[4];
  {
    const v16h ah0 = load_a_lds(sBw, CH, 0, lane);
    const v16h ah1 = load_a_lds(sBw, CH, 1, lane);
#pragma unroll
    for (int t = 0; t < 4; ++t) {
      v8f acc = {};
      acc = wmma_f16(ah0, load_b_gh(ws + PW2_OFF, t, 0, lane), acc);
      acc = wmma_f16(ah1, load_b_gh(ws + PW2_OFF, t, 1, lane), acc);
      const float bias = pos_b2[t * 16 + lo];
#pragma unroll
      for (int r = 0; r < 8; ++r) acc[r] += bias;
      P[t] = acc;
    }
    __builtin_amdgcn_sched_barrier(0);

    // ---- k = feat@wk^T, v = feat@wv^T + pos_enc; energy = q - k + pos_enc -> sB ----
    const v16h af0 = load_a_lds(sFeat[w], CH, 0, lane);
    const v16h af1 = load_a_lds(sFeat[w], CH, 1, lane);
    v8f V[4];
#pragma unroll
    for (int t = 0; t < 4; ++t) {
      v8f kacc = {};
      kacc = wmma_f16(af0, load_b_gh(ws + WK_OFF, t, 0, lane), kacc);
      kacc = wmma_f16(af1, load_b_gh(ws + WK_OFF, t, 1, lane), kacc);
      v8f vacc = {};
      vacc = wmma_f16(af0, load_b_gh(ws + WV_OFF, t, 0, lane), vacc);
      vacc = wmma_f16(af1, load_b_gh(ws + WV_OFF, t, 1, lane), vacc);
#pragma unroll
      for (int r = 0; r < 8; ++r) {
        const float e = Q[t][r] - kacc[r] + P[t][r];
        sBw[(r + 8 * hi) * CH + t * 16 + lo] = (_Float16)e;
        vacc[r] += P[t][r];
      }
      V[t] = vacc;
    }
    __builtin_amdgcn_sched_barrier(0);

    // ---- t1 = relu(energy @ attn_w1^T + b1) (regs), then stage to sB ----
    v8f T1[4];
    {
      const v16h ae0 = load_a_lds(sBw, CH, 0, lane);
      const v16h ae1 = load_a_lds(sBw, CH, 1, lane);
#pragma unroll
      for (int t = 0; t < 4; ++t) {
        v8f acc = {};
        acc = wmma_f16(ae0, load_b_gh(ws + AW1_OFF, t, 0, lane), acc);
        acc = wmma_f16(ae1, load_b_gh(ws + AW1_OFF, t, 1, lane), acc);
        const float bias = attn_b1[t * 16 + lo];
#pragma unroll
        for (int r = 0; r < 8; ++r) acc[r] = fmaxf(acc[r] + bias, 0.0f);
        T1[t] = acc;
      }
    }
    __builtin_amdgcn_sched_barrier(0);
#pragma unroll
    for (int t = 0; t < 4; ++t)
#pragma unroll
      for (int r = 0; r < 8; ++r)
        sBw[(r + 8 * hi) * CH + t * 16 + lo] = (_Float16)T1[t][r];
    __builtin_amdgcn_sched_barrier(0);

    // ---- attn = softmax_k(t1 @ attn_w2^T + b2); out = sum_k attn * v -> sOh ----
    {
      const v16h at0 = load_a_lds(sBw, CH, 0, lane);
      const v16h at1 = load_a_lds(sBw, CH, 1, lane);
#pragma unroll
      for (int t = 0; t < 4; ++t) {
        v8f acc = {};
        acc = wmma_f16(at0, load_b_gh(ws + AW2_OFF, t, 0, lane), acc);
        acc = wmma_f16(at1, load_b_gh(ws + AW2_OFF, t, 1, lane), acc);
        const float bias = attn_b2[t * 16 + lo];
        float m = -1e30f;
#pragma unroll
        for (int r = 0; r < 8; ++r) { acc[r] += bias; m = fmaxf(m, acc[r]); }
        m = fmaxf(m, __shfl_xor(m, 16, 32));   // combine rows 0..7 with 8..15
        float e[8], s = 0.0f;
#pragma unroll
        for (int r = 0; r < 8; ++r) { e[r] = __expf(acc[r] - m); s += e[r]; }
        s += __shfl_xor(s, 16, 32);
        const float inv = 1.0f / s;
        float o = 0.0f;
#pragma unroll
        for (int r = 0; r < 8; ++r) o += e[r] * inv * V[t][r];
        o += __shfl_xor(o, 16, 32);
        if (hi == 0) sOh[w][t * 16 + lo] = (_Float16)o;
      }
    }
  }
  __builtin_amdgcn_sched_barrier(0);

  // ---- y = relu(out @ out_w^T + out_b) via broadcast-A WMMA ----
  {
    const v16h ao0 = load_a_lds(sOh[w], 0, 0, lane);
    const v16h ao1 = load_a_lds(sOh[w], 0, 1, lane);
#pragma unroll
    for (int t = 0; t < 4; ++t) {
      v8f acc = {};
      acc = wmma_f16(ao0, load_b_gh(ws + OW_OFF, t, 0, lane), acc);
      acc = wmma_f16(ao1, load_b_gh(ws + OW_OFF, t, 1, lane), acc);
      if (hi == 0) {
        const int col = t * 16 + lo;
        out[(size_t)p * CH + col] = fmaxf(acc[0] + out_b[col], 0.0f);
      }
    }
  }
}

extern "C" void kernel_launch(void* const* d_in, const int* in_sizes, int n_in,
                              void* d_out, int out_size, void* d_ws, size_t ws_size,
                              hipStream_t stream) {
  const float* x        = (const float*)d_in[0];
  const float* pos      = (const float*)d_in[1];
  const int*   knn_idx  = (const int*)d_in[2];
  const float* base_w   = (const float*)d_in[3];
  const float* spline_w = (const float*)d_in[4];
  const float* scaler   = (const float*)d_in[5];
  const float* pos_w2   = (const float*)d_in[6];
  const float* pos_b2   = (const float*)d_in[7];
  const float* wq       = (const float*)d_in[8];
  const float* wk       = (const float*)d_in[9];
  const float* wv       = (const float*)d_in[10];
  const float* attn_w1  = (const float*)d_in[11];
  const float* attn_b1  = (const float*)d_in[12];
  const float* attn_w2  = (const float*)d_in[13];
  const float* attn_b2  = (const float*)d_in[14];
  const float* out_w    = (const float*)d_in[15];
  const float* out_b    = (const float*)d_in[16];
  float* out = (float*)d_out;
  _Float16* ws = (_Float16*)d_ws;

  hipLaunchKernelGGL(prep_weights, dim3(8), dim3(256), 0, stream,
                     wq, wk, wv, pos_w2, attn_w1, attn_w2, out_w,
                     base_w, spline_w, scaler, ws);
  hipLaunchKernelGGL(kan_knn_transformer, dim3(BN / WAVES), dim3(256), 0, stream,
                     x, pos, knn_idx, pos_b2, attn_b1, attn_b2, out_b, ws, out);
}
